// BiLSTM_CRF_2250562863129
// MI455X (gfx1250) — matrix operand order repaired and checked by reading, not compiled, not run on hardware
//
#include <hip/hip_runtime.h>

// ---------------- problem constants ----------------
#define S_LEN   256
#define BATCH   512
#define EMBD    128
#define HD      128          // per-direction hidden
#define GATES   512          // 4*HD
#define HIDC    256          // concat hidden
#define BSLICE  64           // batch rows per scan workgroup
#define LPAD    64           // labels padded 59 -> 64
#define LREAL   59
#define START_L 57
#define STOP_L  58
#define NEGV    (-10000.0f)
#define LOG2E_F 1.44269504088896340736f
#define LN2_F   0.69314718055994530942f

typedef __bf16 bf16_t;
typedef __attribute__((ext_vector_type(8)))  __bf16 v8bf;
typedef __attribute__((ext_vector_type(16))) __bf16 v16bf;
typedef __attribute__((ext_vector_type(8)))  float  v8f;

// ---- WMMA fragment helpers (ISA 7.12.2 layouts, 16B-aligned vector loads) ----
// A-matrix (16x32 bf16): lane holds row m=lane&15; K = half*8 + {0..7} and 16+half*8+{0..7}
__device__ __forceinline__ v16bf load_fragA(const bf16_t* p) {
  v8bf lo = *(const v8bf*)(p);
  v8bf hi = *(const v8bf*)(p + 16);
  return __builtin_shufflevector(lo, hi, 0,1,2,3,4,5,6,7,8,9,10,11,12,13,14,15);
}
// B-matrix (32x16 bf16): lane holds col n=lane&15; K = half*8 + {0..7} and 16+half*8+{0..7}
__device__ __forceinline__ v16bf load_fragB(const bf16_t* p) {
  v8bf lo = *(const v8bf*)(p);
  v8bf hi = *(const v8bf*)(p + 16);
  return __builtin_shufflevector(lo, hi, 0,1,2,3,4,5,6,7,8,9,10,11,12,13,14,15);
}
__device__ __forceinline__ v8f wmma_bf16(v16bf a, v16bf b, v8f c) {
  return __builtin_amdgcn_wmma_f32_16x16x32_bf16(false, a, false, b, (short)0, c, false, false);
}

// ---- gfx1250 native transcendentals (guarded; fall back to libm) ----
__device__ __forceinline__ float fast_tanh(float x) {
#if __has_builtin(__builtin_amdgcn_tanhf)
  return __builtin_amdgcn_tanhf(x);        // v_tanh_f32
#else
  return tanhf(x);
#endif
}
__device__ __forceinline__ float fast_exp2(float x) {
#if __has_builtin(__builtin_amdgcn_exp2f)
  return __builtin_amdgcn_exp2f(x);        // v_exp_f32
#else
  return exp2f(x);
#endif
}
__device__ __forceinline__ float fast_log2(float x) {
#if __has_builtin(__builtin_amdgcn_logf)
  return __builtin_amdgcn_logf(x);         // v_log_f32
#else
  return log2f(x);
#endif
}
__device__ __forceinline__ float sigm(float x) { return 0.5f * fast_tanh(0.5f * x) + 0.5f; }

// ---------------- kernel 1: weight prep (small tables only) ----------------
__global__ void prep_kernel(const float* __restrict__ bih_f, const float* __restrict__ bhh_f,
                            const float* __restrict__ bih_b, const float* __restrict__ bhh_b,
                            const float* __restrict__ Wout,  const float* __restrict__ bout,
                            const float* __restrict__ trans,
                            bf16_t* __restrict__ Woutb, float* __restrict__ bc,
                            float* __restrict__ boutp, float* __restrict__ transp,
                            float* __restrict__ out) {
  int tid = blockIdx.x * blockDim.x + threadIdx.x;
  int nth = gridDim.x * blockDim.x;
  for (int i = tid; i < LPAD * HIDC; i += nth) {
    int r = i >> 8, k = i & 255;
    Woutb[i] = (r < LREAL) ? (bf16_t)Wout[r * HIDC + k] : (bf16_t)0.0f;
  }
  for (int i = tid; i < GATES; i += nth) {
    bc[i]         = bih_f[i] + bhh_f[i];
    bc[GATES + i] = bih_b[i] + bhh_b[i];
  }
  for (int i = tid; i < LPAD; i += nth)
    boutp[i] = (i < LREAL) ? bout[i] : NEGV;
  for (int i = tid; i < LPAD * LPAD; i += nth) {
    int j = i >> 6, k = i & 63;
    transp[i] = (j < LREAL && k < LREAL) ? trans[j * LREAL + k] : NEGV;
  }
  if (tid == 0) out[0] = 0.0f;
}

// ---------------- kernel 2: embedding gather -> bf16 (S,B,E) ----------------
__global__ void embed_gather_kernel(const int* __restrict__ x,
                                    const float* __restrict__ embed,
                                    bf16_t* __restrict__ embA) {
  size_t idx = (size_t)blockIdx.x * blockDim.x + threadIdx.x;
  if (idx >= (size_t)S_LEN * BATCH * EMBD) return;
  int    e  = (int)(idx & (EMBD - 1));
  size_t sb = idx >> 7;
  int s = (int)(sb >> 9);        // B == 512
  int b = (int)(sb & (BATCH - 1));
  int tok = x[b * S_LEN + s];
  embA[idx] = (bf16_t)embed[(size_t)tok * EMBD + e];
}

// ---------------- kernel 3: LSTM scan ----------------
// Grid: 16 workgroups = 2 directions x 8 batch slices of 64 rows.
// LDS: Wih (128KB) + Whh (128KB) + double-buffered h slice (2x16KB) = 288KB <= 320KB.
// Cell state c lives in VGPRs for the whole scan (fixed wave->tile mapping).
__global__ __launch_bounds__(512) void lstm_scan_kernel(
    const bf16_t* __restrict__ embA,   // (S*B, EMBD) bf16
    const float*  __restrict__ Wih_f,  // (GATES, EMBD) f32
    const float*  __restrict__ Wih_b,
    const float*  __restrict__ Whh_f,  // (GATES, HD) f32
    const float*  __restrict__ Whh_b,
    const float*  __restrict__ bc,     // (2, GATES) f32 (bih+bhh)
    const float*  __restrict__ h0,     // (2, BATCH, HD)
    const float*  __restrict__ c0,     // (2, BATCH, HD)
    bf16_t* __restrict__ hsall)        // (S*B, HIDC) bf16
{
  extern __shared__ __align__(16) bf16_t smem[];
  bf16_t* sWih = smem;                          // GATES*EMBD
  bf16_t* sWhh = smem + GATES * EMBD;           // GATES*HD
  bf16_t* sH   = sWhh + GATES * HD;             // 2 * BSLICE*HD (ping-pong)

  const int dir   = blockIdx.x >> 3;
  const int m0    = (blockIdx.x & 7) * BSLICE;  // batch slice start
  const float* Wih = dir ? Wih_b : Wih_f;
  const float* Whh = dir ? Whh_b : Whh_f;

  // init LDS: both weight matrices -> bf16; h0 slice -> buffer 0
  for (int i = threadIdx.x; i < GATES * EMBD; i += 512) sWih[i] = (bf16_t)Wih[i];
  for (int i = threadIdx.x; i < GATES * HD;   i += 512) sWhh[i] = (bf16_t)Whh[i];
  for (int i = threadIdx.x; i < BSLICE * HD;  i += 512) {
    int m = i >> 7, k = i & (HD - 1);
    sH[i] = (bf16_t)h0[((size_t)dir * BATCH + m0 + m) * HD + k];
  }

  const int wave = threadIdx.x >> 5;
  const int lane = threadIdx.x & 31;
  const int half = lane >> 4;
  const int rowl = lane & 15;
  const float* bcD = bc + dir * GATES;

  // fixed tasks per wave: t = wave*2 + u, t in [0,32): 4 m-tiles x 8 j-tiles
  float cReg[2][8];
  #pragma unroll
  for (int u = 0; u < 2; ++u) {
    const int t  = wave * 2 + u;
    const int mt = (t >> 3) * 16;
    const int n  = (t & 7) * 16 + rowl;
    #pragma unroll
    for (int r = 0; r < 8; ++r)
      cReg[u][r] = c0[((size_t)dir * BATCH + m0 + mt + half * 8 + r) * HD + n];
  }
  __syncthreads();

  for (int s = 0; s < S_LEN; ++s) {
    const int ss  = dir ? (S_LEN - 1 - s) : s;
    const int ssn = dir ? (ss - 1) : (ss + 1);
    bf16_t* sHr = sH + (s & 1) * BSLICE * HD;         // read buffer
    bf16_t* sHw = sH + ((s & 1) ^ 1) * BSLICE * HD;   // write buffer

    #pragma unroll
    for (int u = 0; u < 2; ++u) {
      const int t  = wave * 2 + u;
      const int mt = (t >> 3) * 16;                    // local batch tile
      const int jt = (t & 7) * 16;                     // hidden-col tile
      const int n  = jt + rowl;

      const bf16_t* embp = embA + ((size_t)ss * BATCH + m0 + mt + rowl) * EMBD + half * 8;
      if (s + 1 < S_LEN)  // prefetch next step's A tile (global_prefetch_b8)
        __builtin_prefetch(embA + ((size_t)ssn * BATCH + m0 + mt + rowl) * EMBD, 0, 1);
      const bf16_t* hp = sHr + (mt + rowl) * HD + half * 8;

      v8f acc[4] = {};
      #pragma unroll 1
      for (int kt = 0; kt < 4; ++kt) {
        v16bf aE = load_fragA(embp + kt * 32);
        v16bf aH = load_fragA(hp   + kt * 32);
        #pragma unroll
        for (int g = 0; g < 4; ++g) {
          const int wrow = g * HD + jt + rowl;
          v16bf bI = load_fragB(sWih + wrow * EMBD + half * 8 + kt * 32);
          acc[g] = wmma_bf16(aE, bI, acc[g]);
          v16bf bH = load_fragB(sWhh + wrow * HD + half * 8 + kt * 32);
          acc[g] = wmma_bf16(aH, bH, acc[g]);
        }
      }
      // LSTM cell, accumulator layout: lane -> column n, vgpr r -> row m
      const float b_i = bcD[0 * HD + n];
      const float b_f = bcD[1 * HD + n];
      const float b_g = bcD[2 * HD + n];
      const float b_o = bcD[3 * HD + n];
      #pragma unroll
      for (int r = 0; r < 8; ++r) {
        const int ml = mt + half * 8 + r;              // local batch row
        float iv = sigm(acc[0][r] + b_i);
        float fv = sigm(acc[1][r] + b_f);
        float gv = fast_tanh(acc[2][r] + b_g);
        float ov = sigm(acc[3][r] + b_o);
        float cnew = fv * cReg[u][r] + iv * gv;
        float hv   = ov * fast_tanh(cnew);
        cReg[u][r] = cnew;
        bf16_t hb = (bf16_t)hv;
        sHw[ml * HD + n] = hb;
        hsall[((size_t)ss * BATCH + m0 + ml) * HIDC + dir * HD + n] = hb;
      }
    }
    __syncthreads();
  }
}

// ---------------- kernel 4: emission GEMM (M=131072, N=64, K=256) ----------------
__global__ __launch_bounds__(256) void feats_gemm_kernel(
    const bf16_t* __restrict__ hsall,   // (S*B, HIDC)
    const bf16_t* __restrict__ Woutb,   // (LPAD, HIDC)
    const float*  __restrict__ boutp,   // (LPAD)
    float* __restrict__ feats)          // (S*B, LPAD)
{
  const int gw   = (blockIdx.x * 256 + threadIdx.x) >> 5;   // 0..8191
  const int lane = threadIdx.x & 31;
  const int half = lane >> 4;
  const int rowl = lane & 15;
  const int mt   = gw * 16;

  v8f acc[4] = {};
  const bf16_t* ap = hsall + ((size_t)mt + rowl) * HIDC + half * 8;
  #pragma unroll 1
  for (int kt = 0; kt < 8; ++kt) {
    v16bf a = load_fragA(ap + kt * 32);
    #pragma unroll
    for (int j = 0; j < 4; ++j) {
      v16bf b = load_fragB(Woutb + (size_t)(j * 16 + rowl) * HIDC + half * 8 + kt * 32);
      acc[j] = wmma_bf16(a, b, acc[j]);
    }
  }
  #pragma unroll
  for (int j = 0; j < 4; ++j) {
    const float bv = boutp[j * 16 + rowl];
    #pragma unroll
    for (int r = 0; r < 8; ++r)
      feats[((size_t)mt + half * 8 + r) * LPAD + j * 16 + rowl] = acc[j][r] + bv;
  }
}

// ---------------- kernel 5: CRF forward + gold score ----------------
__global__ __launch_bounds__(64) void crf_kernel(const float* __restrict__ feats,
                                                 const float* __restrict__ transp,
                                                 const int* __restrict__ y,
                                                 float* __restrict__ out) {
  const int b = blockIdx.x;
  const int j = threadIdx.x;          // 0..63
  __shared__ float sT[LPAD * LPAD];
  __shared__ float sA[2][LPAD];
  __shared__ float sRed[LPAD];
  __shared__ float sG[LPAD];

  for (int i = j; i < LPAD * LPAD; i += LPAD) sT[i] = transp[i];
  sA[0][j] = (j == START_L) ? 0.0f : NEGV;
  __syncthreads();

  int cur = 0;
  const float* Tj = &sT[j * LPAD];
  for (int s = 0; s < S_LEN; ++s) {
    const float* ar = sA[cur];
    float mx = -3.0e38f;
    #pragma unroll 8
    for (int i = 0; i < LPAD; ++i) mx = fmaxf(mx, ar[i] + Tj[i]);
    float sum = 0.0f;
    #pragma unroll 8
    for (int i = 0; i < LPAD; ++i) sum += fast_exp2((ar[i] + Tj[i] - mx) * LOG2E_F);
    float ft = feats[((size_t)s * BATCH + b) * LPAD + j];
    sA[cur ^ 1][j] = ft + mx + fast_log2(sum) * LN2_F;
    __syncthreads();
    cur ^= 1;
  }
  sRed[j] = sA[cur][j] + sT[STOP_L * LPAD + j];

  // gold path partial sums (strided over s)
  float gp = 0.0f;
  for (int s = j; s < S_LEN; s += LPAD) {
    int ys = y[b * S_LEN + s];
    int yp = (s == 0) ? START_L : y[b * S_LEN + s - 1];
    gp += feats[((size_t)s * BATCH + b) * LPAD + ys];
    gp += sT[ys * LPAD + yp];
  }
  sG[j] = gp;
  __syncthreads();

  if (j == 0) {
    float mx = -3.0e38f;
    for (int i = 0; i < LPAD; ++i) mx = fmaxf(mx, sRed[i]);
    float sum = 0.0f;
    for (int i = 0; i < LPAD; ++i) sum += fast_exp2((sRed[i] - mx) * LOG2E_F);
    float fwd = mx + fast_log2(sum) * LN2_F;
    float gold = 0.0f;
    for (int i = 0; i < LPAD; ++i) gold += sG[i];
    gold += sT[STOP_L * LPAD + y[b * S_LEN + S_LEN - 1]];
    atomicAdd(out, fwd - gold);
  }
}

// ---------------- host launcher ----------------
extern "C" void kernel_launch(void* const* d_in, const int* in_sizes, int n_in,
                              void* d_out, int out_size, void* d_ws, size_t ws_size,
                              hipStream_t stream) {
  const int*   x     = (const int*)d_in[0];
  const int*   y     = (const int*)d_in[1];
  const float* embed = (const float*)d_in[2];
  const float* Wih_f = (const float*)d_in[3];
  const float* Whh_f = (const float*)d_in[4];
  const float* bih_f = (const float*)d_in[5];
  const float* bhh_f = (const float*)d_in[6];
  const float* Wih_b = (const float*)d_in[7];
  const float* Whh_b = (const float*)d_in[8];
  const float* bih_b = (const float*)d_in[9];
  const float* bhh_b = (const float*)d_in[10];
  const float* Wout  = (const float*)d_in[11];
  const float* bout  = (const float*)d_in[12];
  const float* trans = (const float*)d_in[13];
  const float* h0    = (const float*)d_in[14];
  const float* c0    = (const float*)d_in[15];
  float* out = (float*)d_out;

  // workspace carve-up (256B aligned)
  char* w = (char*)d_ws;
  size_t off = 0;
  auto carve = [&](size_t bytes) { char* p = w + off; off = (off + bytes + 255) & ~(size_t)255; return p; };
  bf16_t* embA   = (bf16_t*)carve((size_t)S_LEN * BATCH * EMBD * sizeof(bf16_t)); // 32 MB
  bf16_t* Woutb  = (bf16_t*)carve((size_t)LPAD * HIDC * sizeof(bf16_t));
  float*  bc     = (float*) carve((size_t)2 * GATES * sizeof(float));
  float*  boutp  = (float*) carve((size_t)LPAD * sizeof(float));
  float*  transp = (float*) carve((size_t)LPAD * LPAD * sizeof(float));
  bf16_t* hsall  = (bf16_t*)carve((size_t)S_LEN * BATCH * HIDC * sizeof(bf16_t)); // 64 MB
  float*  feats  = (float*) carve((size_t)S_LEN * BATCH * LPAD * sizeof(float));  // 32 MB
  (void)ws_size; (void)in_sizes; (void)n_in; (void)out_size;

  prep_kernel<<<64, 256, 0, stream>>>(bih_f, bhh_f, bih_b, bhh_b,
                                      Wout, bout, trans,
                                      Woutb, bc, boutp, transp, out);

  embed_gather_kernel<<<(S_LEN * BATCH * EMBD) / 256, 256, 0, stream>>>(x, embed, embA);

  const size_t scan_lds =
      (size_t)(GATES * EMBD + GATES * HD + 2 * BSLICE * HD) * sizeof(bf16_t); // 288 KB
  lstm_scan_kernel<<<16, 512, scan_lds, stream>>>(embA, Wih_f, Wih_b, Whh_f, Whh_b,
                                                  bc, h0, c0, hsall);

  feats_gemm_kernel<<<(S_LEN * BATCH / 16) / 8, 256, 0, stream>>>(hsall, Woutb, boutp, feats);

  crf_kernel<<<BATCH, LPAD, 0, stream>>>(feats, transp, y, out);
}
